// LightweightMultiDirectionalBlock_18433999634577
// MI455X (gfx1250) — matrix operand order repaired and checked by reading, not compiled, not run on hardware
//
#include <hip/hip_runtime.h>
#include <cstdint>
#include <cstddef>

// ---------- types ----------
typedef __bf16 bf16_t;
typedef bf16_t v16bf __attribute__((ext_vector_type(16)));
typedef bf16_t v8bf  __attribute__((ext_vector_type(8)));
typedef float  v8f   __attribute__((ext_vector_type(8)));
typedef int    v4i_  __attribute__((vector_size(16)));

// address-space-qualified pointers for the async-to-LDS builtin
typedef __attribute__((address_space(1))) v4i_* as1_v4i_p;  // global
typedef __attribute__((address_space(3))) v4i_* as3_v4i_p;  // LDS

// ---------- problem constants ----------
#define B_      4
#define L_      4096
#define DM_     256
#define D_      512
#define NS_     16          // SSM state size N
#define MTOK    (B_ * L_)   // 16384 tokens

#define ACT_NONE     0
#define ACT_GELU     1
#define ACT_SOFTPLUS 2

// LDS W-tile row pitch (elements): 64 data + 8 pad -> 144B rows, avoids the
// 128B-row 2-bank aliasing on the strided 16-bit B reads.
#define WPITCH  72

// ---------------------------------------------------------------------
// fragment load helpers
// ---------------------------------------------------------------------
// A fragment (16-bit A layout): lane 0-15 K{0-7,16-23}, lane 16-31 K{8-15,24-31}
__device__ __forceinline__ v16bf load_afrag(const bf16_t* p) {
    v8bf lo = *reinterpret_cast<const v8bf*>(p);
    v8bf hi = *reinterpret_cast<const v8bf*>(p + 16);
    return __builtin_shufflevector(lo, hi,
        0,1,2,3,4,5,6,7,8,9,10,11,12,13,14,15);
}
// B fragment: lane owns N column, lane 0-15 K{0-7,16-23}, lane 16-31 K{8-15,24-31} (stride ld)
__device__ __forceinline__ v16bf load_bfrag(const bf16_t* p, int ld) {
    v16bf r;
    #pragma unroll
    for (int s = 0; s < 8; ++s) {
        r[s]     = p[(size_t)s * ld];
        r[s + 8] = p[(size_t)(16 + s) * ld];
    }
    return r;
}

// ---------------------------------------------------------------------
// CDNA5 async global->LDS staging (ASYNCcnt-tracked), guarded fallback
// ---------------------------------------------------------------------
#if __has_builtin(__builtin_amdgcn_global_load_async_to_lds_b128)
#define HAS_ASYNC_LDS 1
#else
#define HAS_ASYNC_LDS 0
#endif

__device__ __forceinline__ void stage_b128(const bf16_t* g, bf16_t* l) {
#if HAS_ASYNC_LDS
    __builtin_amdgcn_global_load_async_to_lds_b128(
        (as1_v4i_p)(unsigned long long)(uintptr_t)g,
        (as3_v4i_p)(unsigned)(uintptr_t)l,
        0, 0);
#else
    *reinterpret_cast<v8bf*>(l) = *reinterpret_cast<const v8bf*>(g);
#endif
}

__device__ __forceinline__ void wait_stage() {
#if HAS_ASYNC_LDS
#if __has_builtin(__builtin_amdgcn_s_wait_asynccnt)
    __builtin_amdgcn_s_wait_asynccnt(0);
#else
    asm volatile("s_wait_asynccnt 0x0" ::: "memory");
#endif
#endif
}

// =====================================================================
// Main bf16 WMMA GEMM: out[M,N] = act(A[M,K] @ W[K,N] + bias) (+res)
// Block = 256 threads (8 waves). Wave tile = 32(M) x 64(N).
// Block tile = 256(M) x 64(N). W tile (32x64) staged to LDS (double
// buffered) with async-to-LDS; B fragments read as strided 16-bit LDS loads.
// grid = (M/256, N/64)
// =====================================================================
template<int ACT, bool RES, bool OUTBF>
__global__ __launch_bounds__(256) void gemm_wmma_lds(
    const bf16_t* __restrict__ A, const bf16_t* __restrict__ W,
    const float* __restrict__ bias, const float* __restrict__ res,
    void* __restrict__ out, int M, int N, int K)
{
    __shared__ __align__(16) bf16_t wtile[2][32 * WPITCH];

    const int tid  = threadIdx.x;
    const int lane = tid & 31;
    const int wave = tid >> 5;
    const int m0   = (blockIdx.x * 8 + wave) * 32;
    const int n0   = blockIdx.y * 64;
    const int half = lane >> 4;
    const int mrow = m0 + (lane & 15);

    // ---- staging role: 256 threads move the 32x64 tile (one b128 each)
    const int srow   = tid >> 3;        // 0..31 (K row within tile)
    const int schunk = tid & 7;         // 0..7  (16B chunk within row)
    const bf16_t* gstage = W + (size_t)srow * N + n0 + schunk * 8;
    bf16_t* sdst0 = &wtile[0][srow * WPITCH + schunk * 8];
    bf16_t* sdst1 = &wtile[1][srow * WPITCH + schunk * 8];

    stage_b128(gstage, sdst0);          // k-block 0 -> buffer 0

    // ---- A pointers: rows mrow and mrow+16
    const bf16_t* aptr0 = A + (size_t)mrow * K + half * 8;
    const bf16_t* aptr1 = aptr0 + (size_t)16 * K;
    v16bf a0 = load_afrag(aptr0);
    v16bf a1 = load_afrag(aptr1);

    v8f acc[2][4] = {};
    int p = 0;
    for (int k0 = 0; k0 < K; k0 += 32) {
        wait_stage();
        __syncthreads();                 // buf p ready; prior reads of buf p^1 done
        const bool more = (k0 + 32) < K;
        if (more)
            stage_b128(gstage + (size_t)(k0 + 32) * N, p ? sdst0 : sdst1);

        v16bf a0c = a0, a1c = a1;
        if (more) {                      // prefetch next A fragments into regs
            a0 = load_afrag(aptr0 + k0 + 32);
            a1 = load_afrag(aptr1 + k0 + 32);
        }

        const bf16_t* brow = &wtile[p][(half * 8) * WPITCH + (lane & 15)];
        #pragma unroll
        for (int j = 0; j < 4; ++j) {
            v16bf bfrag = load_bfrag(brow + j * 16, WPITCH);
            acc[0][j] = __builtin_amdgcn_wmma_f32_16x16x32_bf16(
                false, a0c, false, bfrag, (short)0, acc[0][j], false, false);
            acc[1][j] = __builtin_amdgcn_wmma_f32_16x16x32_bf16(
                false, a1c, false, bfrag, (short)0, acc[1][j], false, false);
        }
        p ^= 1;
    }

    // ---- epilogue: C/D layout = VGPR r -> M = r + 8*half, N = lane&15
    const int col = lane & 15;
    #pragma unroll
    for (int i = 0; i < 2; ++i) {
        #pragma unroll
        for (int j = 0; j < 4; ++j) {
            const int n  = n0 + j * 16 + col;
            const float bv = bias ? bias[n] : 0.0f;
            #pragma unroll
            for (int r = 0; r < 8; ++r) {
                const int row = m0 + i * 16 + r + 8 * half;
                float v = acc[i][j][r] + bv;
                if (ACT == ACT_GELU) {
                    v = 0.5f * v * (1.0f + erff(v * 0.70710678118654752f));
                } else if (ACT == ACT_SOFTPLUS) {
                    v = fmaxf(v, 0.0f) + log1pf(__expf(-fabsf(v)));
                }
                const size_t idx = (size_t)row * N + n;
                if (RES) v += res[idx];
                if (OUTBF) ((bf16_t*)out)[idx] = (bf16_t)v;
                else       ((float*)out)[idx]  = v;
            }
        }
    }
}

// =====================================================================
// Small-N GEMM (for the 512->32 B/C projection): wave tile 16x32,
// direct global loads (cost is negligible vs the big GEMMs).
// grid = (M/128, 1), block = 256
// =====================================================================
__global__ __launch_bounds__(256) void gemm_wmma_small(
    const bf16_t* __restrict__ A, const bf16_t* __restrict__ W,
    float* __restrict__ out, int M, int N, int K)
{
    const int lane = threadIdx.x & 31;
    const int wave = threadIdx.x >> 5;
    const int m0   = (blockIdx.x * 8 + wave) * 16;
    const int half = lane >> 4;
    const int mrow = m0 + (lane & 15);

    v8f acc[2] = {};
    const bf16_t* aptr = A + (size_t)mrow * K + half * 8;
    const bf16_t* wptr = W + (size_t)(half * 8) * N + (lane & 15);

    for (int k0 = 0; k0 < K; k0 += 32) {
        v16bf afrag = load_afrag(aptr);
        aptr += 32;
        const bf16_t* wrow = wptr + (size_t)k0 * N;
        #pragma unroll
        for (int j = 0; j < 2; ++j) {
            v16bf bfrag = load_bfrag(wrow + j * 16, N);
            acc[j] = __builtin_amdgcn_wmma_f32_16x16x32_bf16(
                false, afrag, false, bfrag, (short)0, acc[j], false, false);
        }
    }
    const int col = lane & 15;
    #pragma unroll
    for (int j = 0; j < 2; ++j) {
        #pragma unroll
        for (int r = 0; r < 8; ++r) {
            const int row = m0 + r + 8 * half;
            out[(size_t)row * N + j * 16 + col] = acc[j][r];
        }
    }
}

// =====================================================================
// f32 -> bf16 conversion
// =====================================================================
__global__ __launch_bounds__(256) void cvt_bf16(const float* __restrict__ src,
                                                bf16_t* __restrict__ dst, int n)
{
    int i = blockIdx.x * 256 + threadIdx.x;
    if (i < n) dst[i] = (bf16_t)src[i];
}

// Combine w_B[512,16] and w_C[512,16] into Wbc[512,32] (bf16)
__global__ __launch_bounds__(256) void build_wbc(const float* __restrict__ wB,
                                                 const float* __restrict__ wC,
                                                 bf16_t* __restrict__ wbc)
{
    int i = blockIdx.x * 256 + threadIdx.x;
    if (i >= D_ * 32) return;
    int k = i >> 5, n = i & 31;
    float v = (n < NS_) ? wB[k * NS_ + n] : wC[k * NS_ + (n - NS_)];
    wbc[i] = (bf16_t)v;
}

// =====================================================================
// Exact argsort of the 4 scan-order keys via rank counting (keys unique).
// perms[o][rank(i)] = i  <=>  perms[o] = argsort(keys_o)
// =====================================================================
__global__ __launch_bounds__(256) void perm_kernel(const int* __restrict__ coords,
                                                   int* __restrict__ perms)
{
    __shared__ int skey[L_];
    const int o = blockIdx.y;
    for (int j = threadIdx.x; j < L_; j += 256) {
        int r = coords[2 * j], c = coords[2 * j + 1];
        int key;
        switch (o) {
            case 0:  key = r * L_ + c;       break;  // row-major
            case 1:  key = c * L_ + r;       break;  // column-major
            case 2:  key = (r + c) * L_ + r; break;  // diagonal
            default: key = (r - c) * L_ + r; break;  // antidiagonal
        }
        skey[j] = key;
    }
    __syncthreads();
    const int i  = blockIdx.x * 256 + threadIdx.x;
    const int ki = skey[i];
    int rank = 0;
    for (int j = 0; j < L_; ++j) rank += (skey[j] < ki) ? 1 : 0;
    perms[o * L_ + rank] = i;
}

// =====================================================================
// Selective scan. One thread owns one d-channel of one (batch, order):
// h[16] state held in VGPRs; 4096 sequential steps; output scattered
// through perm (fusing the inverse permutation).
// grid = (D/128, B, 4), block = 128
// =====================================================================
__global__ __launch_bounds__(128) void scan_kernel(
    const bf16_t* __restrict__ xp, const float* __restrict__ delta,
    const float* __restrict__ BC, const float* __restrict__ A_log,
    const float* __restrict__ D_ssm, const int* __restrict__ perms,
    bf16_t* __restrict__ ycat)
{
    const int d = blockIdx.x * 128 + threadIdx.x;
    const int b = blockIdx.y;
    const int o = blockIdx.z;
    const int* perm = perms + o * L_;

    float Arow[NS_];
    #pragma unroll
    for (int n = 0; n < NS_; ++n) Arow[n] = -__expf(A_log[d * NS_ + n]);
    const float Dd = D_ssm[d];

    float h[NS_];
    #pragma unroll
    for (int n = 0; n < NS_; ++n) h[n] = 0.0f;

    for (int t = 0; t < L_; ++t) {
        const int tok     = perm[t];                 // block-uniform
        const size_t base = (size_t)b * L_ + tok;
        const float dt = delta[base * D_ + d];
        const float xt = (float)xp[base * D_ + d];
        const float dx = dt * xt;
        const float* bc = BC + base * 32;            // [0:16)=B_t, [16:32)=C_t
        float y = 0.0f;
        #pragma unroll
        for (int n = 0; n < NS_; ++n) {
            const float e = __expf(dt * Arow[n]);
            h[n] = e * h[n] + dx * bc[n];
            y    = fmaf(h[n], bc[NS_ + n], y);
        }
        ycat[base * (4 * D_) + o * D_ + d] = (bf16_t)(y + Dd * xt);
    }
}

// =====================================================================
// LayerNorm over DM=256, one wave per row, wave32 shuffle reduction.
// block = 256 (8 waves), grid = MTOK/8
// =====================================================================
__global__ __launch_bounds__(256) void ln_kernel(const float* __restrict__ x2,
                                                 const float* __restrict__ g,
                                                 const float* __restrict__ bta,
                                                 bf16_t* __restrict__ out)
{
    const int lane = threadIdx.x & 31;
    const int wave = threadIdx.x >> 5;
    const int row  = blockIdx.x * 8 + wave;
    const float* xr = x2 + (size_t)row * DM_;

    float v[8], s = 0.0f, s2 = 0.0f;
    #pragma unroll
    for (int i = 0; i < 8; ++i) {
        v[i] = xr[lane + 32 * i];
        s  += v[i];
        s2 += v[i] * v[i];
    }
    #pragma unroll
    for (int off = 16; off > 0; off >>= 1) {
        s  += __shfl_xor(s,  off, 32);
        s2 += __shfl_xor(s2, off, 32);
    }
    const float m   = s * (1.0f / DM_);
    const float var = s2 * (1.0f / DM_) - m * m;
    const float rs  = rsqrtf(var + 1e-5f);
    #pragma unroll
    for (int i = 0; i < 8; ++i) {
        const int c = lane + 32 * i;
        out[(size_t)row * DM_ + c] = (bf16_t)((v[i] - m) * rs * g[c] + bta[c]);
    }
}

// =====================================================================
// Host orchestration
// =====================================================================
extern "C" void kernel_launch(void* const* d_in, const int* in_sizes, int n_in,
                              void* d_out, int out_size, void* d_ws, size_t ws_size,
                              hipStream_t stream)
{
    (void)in_sizes; (void)n_in; (void)out_size; (void)ws_size;

    const float* x       = (const float*)d_in[0];
    const int*   coords  = (const int*)  d_in[1];
    const float* w_in    = (const float*)d_in[2];
    const float* b_in    = (const float*)d_in[3];
    const float* w_delta = (const float*)d_in[4];
    const float* b_delta = (const float*)d_in[5];
    const float* w_Bm    = (const float*)d_in[6];
    const float* w_Cm    = (const float*)d_in[7];
    const float* A_log   = (const float*)d_in[8];
    const float* D_ssm   = (const float*)d_in[9];
    const float* w_f1    = (const float*)d_in[10];
    const float* b_f1    = (const float*)d_in[11];
    const float* w_f2    = (const float*)d_in[12];
    const float* b_f2    = (const float*)d_in[13];
    const float* w_out   = (const float*)d_in[14];
    const float* b_out   = (const float*)d_in[15];
    const float* ln_g    = (const float*)d_in[16];
    const float* ln_b    = (const float*)d_in[17];
    const float* w_m1    = (const float*)d_in[18];
    const float* b_m1    = (const float*)d_in[19];
    const float* w_m2    = (const float*)d_in[20];
    const float* b_m2    = (const float*)d_in[21];
    float* out = (float*)d_out;

    // ---------------- workspace layout ----------------
    char* ws = (char*)d_ws;
    size_t off = 0;
    auto alloc = [&](size_t bytes) -> void* {
        void* p = ws + off;
        off = (off + bytes + 255) & ~(size_t)255;
        return p;
    };
    bf16_t* x_bf   = (bf16_t*)alloc((size_t)MTOK * DM_ * 2);   //  8.4 MB
    bf16_t* xp_bf  = (bf16_t*)alloc((size_t)MTOK * D_  * 2);   // 16.8 MB
    float*  delta  = (float*) alloc((size_t)MTOK * D_  * 4);   // 33.6 MB
    float*  BC     = (float*) alloc((size_t)MTOK * 32  * 4);   //  2.1 MB
    char*   bigreg = (char*)  alloc((size_t)MTOK * 4 * D_ * 2);// 67.1 MB (ycat; reused)
    bf16_t* t1_bf  = (bf16_t*)alloc((size_t)MTOK * 1024 * 2);  // 33.6 MB (t1; reused as t2)
    bf16_t* wb_in  = (bf16_t*)alloc((size_t)DM_ * D_   * 2);
    bf16_t* wb_dl  = (bf16_t*)alloc((size_t)D_  * D_   * 2);
    bf16_t* wb_bc  = (bf16_t*)alloc((size_t)D_  * 32   * 2);
    bf16_t* wb_f1  = (bf16_t*)alloc((size_t)4 * D_ * 2 * D_ * 2);
    bf16_t* wb_f2  = (bf16_t*)alloc((size_t)2 * D_ * D_ * 2);
    bf16_t* wb_o   = (bf16_t*)alloc((size_t)D_ * DM_   * 2);
    bf16_t* wb_m1  = (bf16_t*)alloc((size_t)DM_ * 4 * DM_ * 2);
    bf16_t* wb_m2  = (bf16_t*)alloc((size_t)4 * DM_ * DM_ * 2);
    int*    perms  = (int*)   alloc((size_t)4 * L_ * 4);

    // phase-overlapped aliases inside bigreg (ycat dead after fusion MLP #1)
    bf16_t* ycat  = (bf16_t*)bigreg;
    bf16_t* yff   = (bf16_t*)bigreg;                               // 16.8 MB
    float*  x2    = (float*)(bigreg + (size_t)MTOK * D_ * 2);      // 16.8 MB
    bf16_t* hln   = (bf16_t*)(bigreg + (size_t)MTOK * D_ * 2
                                     + (size_t)MTOK * DM_ * 4);    //  8.4 MB
    bf16_t* t2_bf = t1_bf;

    const dim3 blk(256);
    auto cvt = [&](const float* s, bf16_t* d, int n) {
        cvt_bf16<<<dim3((n + 255) / 256), blk, 0, stream>>>(s, d, n);
    };

    // ---- 0) conversions (deterministic, re-done every call) ----
    cvt(x,       x_bf,  MTOK * DM_);
    cvt(w_in,    wb_in, DM_ * D_);
    cvt(w_delta, wb_dl, D_ * D_);
    cvt(w_f1,    wb_f1, 4 * D_ * 2 * D_);
    cvt(w_f2,    wb_f2, 2 * D_ * D_);
    cvt(w_out,   wb_o,  D_ * DM_);
    cvt(w_m1,    wb_m1, DM_ * 4 * DM_);
    cvt(w_m2,    wb_m2, 4 * DM_ * DM_);
    build_wbc<<<dim3((D_ * 32 + 255) / 256), blk, 0, stream>>>(w_Bm, w_Cm, wb_bc);

    // ---- 1) scan-order permutations ----
    perm_kernel<<<dim3(L_ / 256, 4), blk, 0, stream>>>(coords, perms);

    // ---- 2) in-projection: xp = x @ w_in + b_in  -> bf16 ----
    gemm_wmma_lds<ACT_NONE, false, true><<<dim3(MTOK / 256, D_ / 64), blk, 0, stream>>>(
        x_bf, wb_in, b_in, nullptr, xp_bf, MTOK, D_, DM_);

    // ---- 3) delta = softplus(xp @ w_delta + b_delta) -> f32 ----
    gemm_wmma_lds<ACT_SOFTPLUS, false, false><<<dim3(MTOK / 256, D_ / 64), blk, 0, stream>>>(
        xp_bf, wb_dl, b_delta, nullptr, delta, MTOK, D_, D_);

    // ---- 4) [Bm | Cm] = xp @ [w_B | w_C] -> f32 [MTOK, 32] ----
    gemm_wmma_small<<<dim3(MTOK / 128, 1), blk, 0, stream>>>(
        xp_bf, wb_bc, BC, MTOK, 32, D_);

    // ---- 5) 4-direction selective scans (permutation fused) ----
    scan_kernel<<<dim3(D_ / 128, B_, 4), dim3(128), 0, stream>>>(
        xp_bf, delta, BC, A_log, D_ssm, perms, ycat);

    // ---- 6) fusion MLP: t1 = gelu(ycat @ w_f1 + b_f1) ----
    gemm_wmma_lds<ACT_GELU, false, true><<<dim3(MTOK / 256, (2 * D_) / 64), blk, 0, stream>>>(
        ycat, wb_f1, b_f1, nullptr, t1_bf, MTOK, 2 * D_, 4 * D_);

    // ---- 7) yff = t1 @ w_f2 + b_f2 ----
    gemm_wmma_lds<ACT_NONE, false, true><<<dim3(MTOK / 256, D_ / 64), blk, 0, stream>>>(
        t1_bf, wb_f2, b_f2, nullptr, yff, MTOK, D_, 2 * D_);

    // ---- 8) x2 = yff @ w_out + b_out + x ----
    gemm_wmma_lds<ACT_NONE, true, false><<<dim3(MTOK / 256, DM_ / 64), blk, 0, stream>>>(
        yff, wb_o, b_out, x, x2, MTOK, DM_, D_);

    // ---- 9) hln = LN(x2) ----
    ln_kernel<<<dim3(MTOK / 8), blk, 0, stream>>>(x2, ln_g, ln_b, hln);

    // ---- 10) t2 = gelu(hln @ w_m1 + b_m1) ----
    gemm_wmma_lds<ACT_GELU, false, true><<<dim3(MTOK / 256, (4 * DM_) / 64), blk, 0, stream>>>(
        hln, wb_m1, b_m1, nullptr, t2_bf, MTOK, 4 * DM_, DM_);

    // ---- 11) out = t2 @ w_m2 + b_m2 + x2 ----
    gemm_wmma_lds<ACT_NONE, true, false><<<dim3(MTOK / 256, DM_ / 64), blk, 0, stream>>>(
        t2_bf, wb_m2, b_m2, x2, out, MTOK, DM_, 4 * DM_);
}
